// R_GCN_46231027974524
// MI455X (gfx1250) — matrix operand order repaired and checked by reading, not compiled, not run on hardware
//
#include <hip/hip_runtime.h>
#include <hip/hip_bf16.h>

typedef __attribute__((ext_vector_type(16))) _Float16 v16h;
typedef __attribute__((ext_vector_type(8)))  float    v8f;

#define N_NODES 100000
#define N_REL   100
#define N_EDGES 1000000
#define N_DEC   200000

// LDS staging for fc_w (f16): row stride 784 halves = 392 dwords; 392%64==8 so
// consecutive lanes' b128 reads land on disjoint 8-bank groups (2-way minimum).
#define ENC_LDS_STRIDE 784
#define ENC_LDS_BYTES  (64 * ENC_LDS_STRIDE * 2)   // 100352 B (< 320KB WGP LDS)

// ---------------------------------------------------------------------------
// Utility: zero-fill (graph-capture-safe replacement for memset)
// ---------------------------------------------------------------------------
__global__ void k_fill_zero(float* __restrict__ p, size_t n) {
    size_t i = (size_t)blockIdx.x * blockDim.x + threadIdx.x;
    if (i < n) p[i] = 0.0f;
}

// One-shot: fc_w f32 -> f16 (48K elements; lives in workspace, L2-resident)
__global__ void k_convert_w(const float* __restrict__ w, _Float16* __restrict__ w16,
                            int n) {
    int i = blockIdx.x * blockDim.x + threadIdx.x;
    if (i < n) w16[i] = (_Float16)w[i];
}

// ---------------------------------------------------------------------------
// Per-(dst, rel) edge counting + per-edge mean normalization (shared by layers)
// ---------------------------------------------------------------------------
__global__ void k_count_edges(const int* __restrict__ dst,
                              const int* __restrict__ etype,
                              float* __restrict__ cnt, int n_edges) {
    int e = blockIdx.x * blockDim.x + threadIdx.x;
    if (e >= n_edges) return;
    atomicAdd(&cnt[(size_t)dst[e] * N_REL + etype[e]], 1.0f);
}

__global__ void k_compute_norm(const int* __restrict__ dst,
                               const int* __restrict__ etype,
                               const float* __restrict__ cnt,
                               float* __restrict__ norm, int n_edges) {
    int e = blockIdx.x * blockDim.x + threadIdx.x;
    if (e >= n_edges) return;
    float c = cnt[(size_t)dst[e] * N_REL + etype[e]];
    norm[e] = 1.0f / fmaxf(c, 1.0f);
}

// ---------------------------------------------------------------------------
// Encoder GEMM via WMMA: h[n,o] = sum_k node_emb[n,k]*fc_w[o,k] + fc_b[o] + rand[n,o]
// B (fc_w, f16) staged in LDS once per block; A streamed from HBM with on-the-fly
// f16 conversion. One wave per 16-row block; 4 output tiles; K stepped by 32.
// A/B use contiguous k = 16*hi + i (same in both, not ISA table order); C/D per CDNA5 ISA 7.12.2.
// ---------------------------------------------------------------------------
__global__ void k_encoder_wmma(const float* __restrict__ node_emb,   // [N,768]
                               const _Float16* __restrict__ fcw16,   // [64,768] f16
                               const float* __restrict__ fc_b,       // [64]
                               const float* __restrict__ rand_embed, // [N,64]
                               float* __restrict__ h, int n_nodes) {
    extern __shared__ _Float16 lds_w[];   // [64][ENC_LDS_STRIDE]

    // ---- cooperative LDS fill: 64 rows x 96 chunks of 8 f16 (16B each)
    for (int c = threadIdx.x; c < 64 * 96; c += blockDim.x) {
        int r = c / 96, cc = c % 96;
        *(uint4*)(lds_w + r * ENC_LDS_STRIDE + cc * 8) =
            *(const uint4*)(fcw16 + r * 768 + cc * 8);
    }
    __syncthreads();   // all waves participate before any tail-wave exit

    const int lane = threadIdx.x & 31;
    const int wave = threadIdx.x >> 5;
    const int rowBlock = blockIdx.x * (blockDim.x >> 5) + wave;  // 16 rows/wave
    if (rowBlock * 16 >= n_nodes) return;                        // wave-uniform

    const int row = lane & 15;   // M for A loads / N for B & D
    const int hi  = lane >> 4;   // half-wave selector
    const size_t m = (size_t)(rowBlock * 16 + row);

    v8f acc[4];
    #pragma unroll
    for (int t = 0; t < 4; ++t) acc[t] = v8f{};

    const float* arowbase = node_emb + m * 768 + hi * 16;
    for (int k0 = 0; k0 < 768; k0 += 32) {
        // A fragment (16x32 f16): lanes 0-15 K{0..15}, lanes 16-31 K{16..31}
        const float* arow = arowbase + k0;
        __builtin_prefetch(arow + 32, 0, 1);   // next K tile -> global_prefetch_b8
        v16h a;
        #pragma unroll
        for (int i = 0; i < 8; ++i) a[i]     = (_Float16)arow[i];
        #pragma unroll
        for (int i = 0; i < 8; ++i) a[i + 8] = (_Float16)arow[8 + i];

        // B fragments from LDS: lane holds column o=t*16+row, K=k0+hi*16+i
        #pragma unroll
        for (int t = 0; t < 4; ++t) {
            v16h b = *(const v16h*)(lds_w + (size_t)(t * 16 + row) * ENC_LDS_STRIDE
                                    + k0 + hi * 16);               // ds_load_b128
            acc[t] = __builtin_amdgcn_wmma_f32_16x16x32_f16(
                false, a, false, b, (short)0, acc[t], false, false);
        }
    }

    // D layout: M = v + 8*hi, N = row
    #pragma unroll
    for (int t = 0; t < 4; ++t) {
        const int o = t * 16 + row;
        const float bo = fc_b[o];
        #pragma unroll
        for (int v = 0; v < 8; ++v) {
            const size_t mo = (size_t)(rowBlock * 16 + v + hi * 8);
            h[mo * 64 + o] = acc[t][v] + bo + rand_embed[mo * 64 + o];
        }
    }
}

// ---------------------------------------------------------------------------
// Basis projection: y_b = x @ basis[b]  (node-level basis trick)
// ---------------------------------------------------------------------------
template <int IN, int SX, int OUT, int SO>
__global__ void k_basis_proj(const float* __restrict__ x,
                             const float* __restrict__ basis,  // [2, IN, OUT]
                             float* __restrict__ y0, float* __restrict__ y1,
                             int n_nodes) {
    int idx = blockIdx.x * blockDim.x + threadIdx.x;
    int node = idx / OUT, o = idx % OUT;
    if (node >= n_nodes) return;
    const float* xr = x + (size_t)node * SX;
    float s0 = 0.0f, s1 = 0.0f;
    #pragma unroll
    for (int k = 0; k < IN; ++k) {
        float xv = xr[k];
        s0 = fmaf(xv, basis[k * OUT + o], s0);
        s1 = fmaf(xv, basis[IN * OUT + k * OUT + o], s1);
    }
    y0[(size_t)node * SO + o] = s0;
    y1[(size_t)node * SO + o] = s1;
}

// ---------------------------------------------------------------------------
// Edge scatter, OUT<=16 variant: TWO edges per wave (lanes 0..15 / 16..31).
// ---------------------------------------------------------------------------
template <int OUT, int SO>
__global__ void k_scatter_msg_pack2(const int* __restrict__ src,
                                    const int* __restrict__ dst,
                                    const int* __restrict__ etype,
                                    const float* __restrict__ comp,
                                    const float* __restrict__ norm,
                                    const float* __restrict__ y0,
                                    const float* __restrict__ y1,
                                    float* __restrict__ agg, int n_edges) {
    int w    = (blockIdx.x * blockDim.x + threadIdx.x) >> 5;
    int lane = threadIdx.x & 31;
    int half = lane >> 4, ch = lane & 15;
    int e = w * 2 + half;
    if (e < n_edges && ch < OUT) {
        int s = src[e], d = dst[e], r = etype[e];
        float c0 = comp[r * 2 + 0], c1 = comp[r * 2 + 1];
        float v = (c0 * y0[(size_t)s * SO + ch] +
                   c1 * y1[(size_t)s * SO + ch]) * norm[e];
        atomicAdd(&agg[(size_t)d * SO + ch], v);
    }
}

// Edge scatter, OUT==32 variant: one wave per edge, lane = channel.
template <int OUT, int SO>
__global__ void k_scatter_msg(const int* __restrict__ src,
                              const int* __restrict__ dst,
                              const int* __restrict__ etype,
                              const float* __restrict__ comp,
                              const float* __restrict__ norm,
                              const float* __restrict__ y0,
                              const float* __restrict__ y1,
                              float* __restrict__ agg, int n_edges) {
    int e    = (blockIdx.x * blockDim.x + threadIdx.x) >> 5;
    int lane = threadIdx.x & 31;
    if (e >= n_edges) return;
    int s = src[e], d = dst[e], r = etype[e];
    float c0 = comp[r * 2 + 0], c1 = comp[r * 2 + 1];
    float nm = norm[e];
    if (lane < OUT) {
        float v = (c0 * y0[(size_t)s * SO + lane] +
                   c1 * y1[(size_t)s * SO + lane]) * nm;
        atomicAdd(&agg[(size_t)d * SO + lane], v);
    }
}

// ---------------------------------------------------------------------------
// Finalize: out = agg + x @ root + bias  (+ optional ReLU)
// ---------------------------------------------------------------------------
template <int IN, int SX, int OUT, int SO, int SOUT, bool RELU>
__global__ void k_finalize(const float* __restrict__ agg,
                           const float* __restrict__ x,
                           const float* __restrict__ root,   // [IN, OUT]
                           const float* __restrict__ bias,   // [OUT]
                           float* __restrict__ out, int n_nodes) {
    int idx = blockIdx.x * blockDim.x + threadIdx.x;
    int node = idx / OUT, o = idx % OUT;
    if (node >= n_nodes) return;
    const float* xr = x + (size_t)node * SX;
    float s = agg[(size_t)node * SO + o] + bias[o];
    #pragma unroll
    for (int k = 0; k < IN; ++k) s = fmaf(xr[k], root[k * OUT + o], s);
    if (RELU) s = fmaxf(s, 0.0f);
    out[(size_t)node * SOUT + o] = s;
}

// ---------------------------------------------------------------------------
// DistMult decode: one wave32 per score; lane-per-channel, shfl_xor reduction.
// ---------------------------------------------------------------------------
__global__ void k_decode(const int* __restrict__ pos_ei,    // [2,P]
                         const int* __restrict__ neg_ei,    // [2,P]
                         const int* __restrict__ dec_type,  // [P]
                         const float* __restrict__ z,       // [N,32]
                         const float* __restrict__ rel_emb, // [R,32]
                         float* __restrict__ out, int P) {
    int i    = (blockIdx.x * blockDim.x + threadIdx.x) >> 5;
    int lane = threadIdx.x & 31;
    if (i >= 2 * P) return;
    int s, d, r;
    if (i < P) { s = pos_ei[i];       d = pos_ei[P + i];       r = dec_type[i]; }
    else       { int j = i - P; s = neg_ei[j]; d = neg_ei[P + j]; r = dec_type[j]; }
    float v = z[(size_t)s * 32 + lane] * rel_emb[(size_t)r * 32 + lane] *
              z[(size_t)d * 32 + lane];
    #pragma unroll
    for (int off = 16; off > 0; off >>= 1) v += __shfl_xor(v, off, 32);
    if (lane == 0) out[i] = v;
}

// ---------------------------------------------------------------------------
// Host-side orchestration (all on `stream`; graph-capture safe)
// ---------------------------------------------------------------------------
extern "C" void kernel_launch(void* const* d_in, const int* in_sizes, int n_in,
                              void* d_out, int out_size, void* d_ws, size_t ws_size,
                              hipStream_t stream) {
    const int* edge_index = (const int*)d_in[0];   // [2,E]
    const int* edge_type  = (const int*)d_in[1];   // [E]
    const int* pos_ei     = (const int*)d_in[2];   // [2,P]
    const int* neg_ei     = (const int*)d_in[3];   // [2,P]
    const int* dec_type   = (const int*)d_in[4];   // [P]
    const float* node_emb   = (const float*)d_in[5];
    const float* rand_embed = (const float*)d_in[6];
    const float* fc_w       = (const float*)d_in[7];
    const float* fc_b       = (const float*)d_in[8];
    const float* c1_basis = (const float*)d_in[9];
    const float* c1_comp  = (const float*)d_in[10];
    const float* c1_root  = (const float*)d_in[11];
    const float* c1_bias  = (const float*)d_in[12];
    const float* c2_basis = (const float*)d_in[13];
    const float* c2_comp  = (const float*)d_in[14];
    const float* c2_root  = (const float*)d_in[15];
    const float* c2_bias  = (const float*)d_in[16];
    const float* c3_basis = (const float*)d_in[17];
    const float* c3_comp  = (const float*)d_in[18];
    const float* c3_root  = (const float*)d_in[19];
    const float* c3_bias  = (const float*)d_in[20];
    const float* rel_emb  = (const float*)d_in[21];

    const int N = N_NODES, E = N_EDGES, P = N_DEC;
    const int* e_src = edge_index;       // row 0
    const int* e_dst = edge_index + E;   // row 1

    // ---- workspace layout (floats). cnt is dead after k_compute_norm, so the
    //      encoder output h aliases it (stream order guarantees safety).
    float* ws = (float*)d_ws;
    size_t off = 0;
    float* cnt  = ws + off; off += (size_t)N * N_REL;   // 10.0M floats
    float* h    = cnt;                                  // 6.4M (aliases cnt)
    float* norm = ws + off; off += (size_t)E;           // 1.0M
    float* x1   = ws + off; off += (size_t)N * 16;      // 1.6M (14 ch, stride 16)
    float* x2   = ws + off; off += (size_t)N * 32;      // 3.2M
    float* zb   = ws + off; off += (size_t)N * 32;      // 3.2M
    float* y0   = ws + off; off += (size_t)N * 32;      // 3.2M
    float* y1   = ws + off; off += (size_t)N * 32;      // 3.2M
    float* agg  = ws + off; off += (size_t)N * 32;      // 3.2M
    _Float16* fcw16 = (_Float16*)(ws + off); off += 64 * 768 / 2;  // 24.6K floats

    const int T = 256;
    auto blk = [](size_t n, int t) { return (unsigned)((n + t - 1) / t); };

    // ---- normalization (computed once, shared by all 3 layers)
    k_fill_zero<<<blk((size_t)N * N_REL, T), T, 0, stream>>>(cnt, (size_t)N * N_REL);
    k_count_edges<<<blk(E, T), T, 0, stream>>>(e_dst, edge_type, cnt, E);
    k_compute_norm<<<blk(E, T), T, 0, stream>>>(e_dst, edge_type, cnt, norm, E);

    // ---- fc_w -> f16 once, then WMMA encoder with LDS-staged B
    k_convert_w<<<blk(64 * 768, T), T, 0, stream>>>(fc_w, fcw16, 64 * 768);
    k_encoder_wmma<<<blk(6250, 8), 256, ENC_LDS_BYTES, stream>>>(
        node_emb, fcw16, fc_b, rand_embed, h, N);

    // ---- layer 1: 64 -> 14 (ReLU); pack-2 scatter (OUT<=16)
    k_basis_proj<64, 64, 14, 16><<<blk((size_t)N * 14, T), T, 0, stream>>>(
        h, c1_basis, y0, y1, N);
    k_fill_zero<<<blk((size_t)N * 16, T), T, 0, stream>>>(agg, (size_t)N * 16);
    k_scatter_msg_pack2<14, 16><<<blk(((size_t)(E + 1) / 2) * 32, T), T, 0, stream>>>(
        e_src, e_dst, edge_type, c1_comp, norm, y0, y1, agg, E);
    k_finalize<64, 64, 14, 16, 16, true><<<blk((size_t)N * 14, T), T, 0, stream>>>(
        agg, h, c1_root, c1_bias, x1, N);

    // ---- layer 2: 14 -> 32
    k_basis_proj<14, 16, 32, 32><<<blk((size_t)N * 32, T), T, 0, stream>>>(
        x1, c2_basis, y0, y1, N);
    k_fill_zero<<<blk((size_t)N * 32, T), T, 0, stream>>>(agg, (size_t)N * 32);
    k_scatter_msg<32, 32><<<blk((size_t)E * 32, T), T, 0, stream>>>(
        e_src, e_dst, edge_type, c2_comp, norm, y0, y1, agg, E);
    k_finalize<14, 16, 32, 32, 32, false><<<blk((size_t)N * 32, T), T, 0, stream>>>(
        agg, x1, c2_root, c2_bias, x2, N);

    // ---- layer 3: 32 -> 32
    k_basis_proj<32, 32, 32, 32><<<blk((size_t)N * 32, T), T, 0, stream>>>(
        x2, c3_basis, y0, y1, N);
    k_fill_zero<<<blk((size_t)N * 32, T), T, 0, stream>>>(agg, (size_t)N * 32);
    k_scatter_msg<32, 32><<<blk((size_t)E * 32, T), T, 0, stream>>>(
        e_src, e_dst, edge_type, c3_comp, norm, y0, y1, agg, E);
    k_finalize<32, 32, 32, 32, 32, false><<<blk((size_t)N * 32, T), T, 0, stream>>>(
        agg, x2, c3_root, c3_bias, zb, N);

    // ---- decode: 2P scores, one wave each
    k_decode<<<blk((size_t)2 * P * 32, T), T, 0, stream>>>(
        pos_ei, neg_ei, dec_type, zb, rel_emb, (float*)d_out, P);
}
